// GRUCell_56959856280321
// MI455X (gfx1250) — matrix operand order repaired and checked by reading, not compiled, not run on hardware
//
#include <hip/hip_runtime.h>
#include <hip/hip_bf16.h>
#include <math.h>

// ---------------- problem constants ----------------
constexpr int M_ = 8192;   // batch
constexpr int N_ = 2048;   // hidden
constexpr int K_ = 2048;   // per-segment K (IN == H == 2048)
constexpr float EPSV = 1e-5f;

typedef __bf16 bf16;
typedef __attribute__((ext_vector_type(16))) __bf16       v16bf;
typedef __attribute__((ext_vector_type(4)))  __bf16       v4bf;
typedef __attribute__((ext_vector_type(8)))  float        v8f;
typedef __attribute__((ext_vector_type(4)))  float        f32x4;
typedef __attribute__((ext_vector_type(4)))  unsigned int u32x4;

union FragBF { v16bf v; u32x4 u[2]; };

// ---------------- fp32 -> bf16 cast (4 elems/thread, 8B vector store) ----------------
__global__ __launch_bounds__(256) void cast_f32_bf16(const float* __restrict__ in,
                                                     bf16* __restrict__ out, int n)
{
    int i = (blockIdx.x * 256 + threadIdx.x) * 4;
    if (i + 3 < n) {
        f32x4 f = *(const f32x4*)(in + i);
        v4bf o;
        o.x = (bf16)f.x; o.y = (bf16)f.y; o.z = (bf16)f.z; o.w = (bf16)f.w;
        *(v4bf*)(out + i) = o;
    }
}

// ---------------- fused dual-K GEMM:  C = A0*B0^T + A1*B1^T + bias (+epilogue) ----------------
// Block tile 256x128, 8 waves (wave32) as 4(M) x 2(N), wave tile 64x64 -> 4x4 WMMA accums.
// K loop: 128 steps of 32 (first 64 over {A0,B0}, next 64 over {A1,B1}).
// Staging: global_load_async_to_lds_b128 into double-buffered LDS, ASYNCcnt-pipelined
// two tiles deep (wait <=6 keeps only the newest tile's 6 asyncs in flight).
// EPI==0 : C = acc + bias
// EPI==1 : C = (1-z)*h_prev + z*tanh(acc + bias)
template<int EPI>
__global__ __launch_bounds__(256) void gru_gemm(
    const bf16* __restrict__ A0, const bf16* __restrict__ B0,
    const bf16* __restrict__ A1, const bf16* __restrict__ B1,
    const float* __restrict__ bias,
    const float* __restrict__ zbuf,
    const float* __restrict__ hprev,
    float* __restrict__ C)
{
    constexpr int LDT = 40;                 // padded LDS row stride (elements)
    __shared__ __align__(16) bf16 sA[2][256 * LDT];   // 2 x 20 KB
    __shared__ __align__(16) bf16 sB[2][128 * LDT];   // 2 x 10 KB

    const int tid   = threadIdx.x;
    const int lane  = tid & 31;
    const int wave  = tid >> 5;
    const int waveM = wave >> 1;            // 0..3  (64-row slabs)
    const int waveN = wave & 1;             // 0..1  (64-col slabs)
    const int kh    = lane >> 4;            // lane half (0/1)
    const int l16   = lane & 15;

    const int mBase = blockIdx.y * 256;
    const int nBase = blockIdx.x * 128;

    // async staging assignments: A tile 256x32 (4 chunks/thread), B tile 128x32 (2 chunks/thread)
    const int arow = tid;                   // A row handled by this thread
    const int brow = tid >> 1;              // B row handled by this thread
    const int bci  = (tid & 1) * 2;         // B first 16B chunk index

    auto stage = [&](int kt, int buf) {
        const bf16* Ab = (kt < 64) ? A0 : A1;
        const bf16* Bb = (kt < 64) ? B0 : B1;
        const int   kk = (kt & 63) * 32;
        const bf16* pa = Ab + (size_t)(mBase + arow) * K_ + kk;
        unsigned la = (unsigned)(size_t)(const void*)&sA[buf][arow * LDT];
        #pragma unroll
        for (int ci = 0; ci < 4; ++ci) {
            unsigned long long ga = (unsigned long long)(pa + ci * 8);
            unsigned ldsa = la + ci * 16;
            asm volatile("global_load_async_to_lds_b128 %0, %1, off"
                         :: "v"(ldsa), "v"(ga) : "memory");
        }
        const bf16* pb = Bb + (size_t)(nBase + brow) * K_ + kk + bci * 8;
        unsigned lb = (unsigned)(size_t)(const void*)&sB[buf][brow * LDT + bci * 8];
        #pragma unroll
        for (int ci = 0; ci < 2; ++ci) {
            unsigned long long gb = (unsigned long long)(pb + ci * 8);
            unsigned ldsb = lb + ci * 16;
            asm volatile("global_load_async_to_lds_b128 %0, %1, off"
                         :: "v"(ldsb), "v"(gb) : "memory");
        }
    };

    v8f acc[4][4] = {};

    stage(0, 0);
    stage(1, 1);

    for (int kt = 0; kt < 128; ++kt) {
        // tile kt complete when only the newest tile's 6 asyncs remain outstanding
        if (kt < 127) asm volatile("s_wait_asynccnt 0x6" ::: "memory");
        else          asm volatile("s_wait_asynccnt 0x0" ::: "memory");
        __syncthreads();                    // publish buf[kt&1] to all waves

        const int buf = kt & 1;
        // A fragments: dense 16-bit 16x32 layout -> lane half kh reads
        // K = kh*8..kh*8+7 and K = 16+kh*8..+7 of its row.
        FragBF a[4];
        #pragma unroll
        for (int mi = 0; mi < 4; ++mi) {
            const bf16* p = &sA[buf][(waveM * 64 + mi * 16 + l16) * LDT + kh * 8];
            a[mi].u[0] = *(const u32x4*)p;
            a[mi].u[1] = *(const u32x4*)(p + 16);
        }
        // B fragments: same K order as A -> lane half kh reads
        // K = kh*8..kh*8+7 and K = 16+kh*8..+7 of its column.
        FragBF b[4];
        #pragma unroll
        for (int ni = 0; ni < 4; ++ni) {
            const bf16* p = &sB[buf][(waveN * 64 + ni * 16 + l16) * LDT + kh * 8];
            b[ni].u[0] = *(const u32x4*)p;
            b[ni].u[1] = *(const u32x4*)(p + 16);
        }

        #pragma unroll
        for (int mi = 0; mi < 4; ++mi)
            #pragma unroll
            for (int ni = 0; ni < 4; ++ni)
                acc[mi][ni] = __builtin_amdgcn_wmma_f32_16x16x32_bf16(
                    false, a[mi].v, false, b[ni].v,
                    (short)0, acc[mi][ni], false, false);

        __syncthreads();                    // all waves done reading buf[kt&1]
        if (kt + 2 < 128) stage(kt + 2, buf);   // refill the drained buffer
    }

    // Epilogue. C/D layout: VGPR r -> M = kh*8 + r ; N = l16.
    #pragma unroll
    for (int mi = 0; mi < 4; ++mi) {
        #pragma unroll
        for (int ni = 0; ni < 4; ++ni) {
            const int col  = nBase + waveN * 64 + ni * 16 + l16;
            const int row0 = mBase + waveM * 64 + mi * 16 + kh * 8;
            const float bv = bias[col];
            #pragma unroll
            for (int r = 0; r < 8; ++r) {
                const size_t idx = (size_t)(row0 + r) * N_ + col;
                float v = acc[mi][ni][r] + bv;
                if (EPI == 0) {
                    C[idx] = v;
                } else {
                    const float zv = zbuf[idx];
                    const float hv = hprev[idx];
                    C[idx] = (1.0f - zv) * hv + zv * tanhf(v);
                }
            }
        }
    }
}

// ---------------- per-row LayerNorm + sigmoid for z and r; emit z (in place) and bf16 r*h ----
__global__ __launch_bounds__(256) void ln_sig_kernel(
    float* __restrict__ preZ,               // in: pre-activation z ; out: z
    const float* __restrict__ preR,
    const float* __restrict__ hprev,
    const float* __restrict__ gz, const float* __restrict__ bz,
    const float* __restrict__ gr, const float* __restrict__ br,
    bf16* __restrict__ rhb)                 // out: bf16(r * h_prev)
{
    const int row = blockIdx.x;
    const int t   = threadIdx.x;
    float*       pz  = preZ  + (size_t)row * N_;
    const float* pr  = preR  + (size_t)row * N_;
    const float* ph  = hprev + (size_t)row * N_;
    bf16*        prh = rhb   + (size_t)row * N_;

    float s0 = 0.f, s1 = 0.f, s2 = 0.f, s3 = 0.f;
    for (int c = t; c < N_; c += 256) {
        float a = pz[c]; s0 += a; s1 += a * a;
        float b = pr[c]; s2 += b; s3 += b * b;
    }
    __shared__ float r0[256], r1[256], r2[256], r3[256];
    r0[t] = s0; r1[t] = s1; r2[t] = s2; r3[t] = s3;
    __syncthreads();
    for (int st = 128; st > 0; st >>= 1) {
        if (t < st) {
            r0[t] += r0[t + st]; r1[t] += r1[t + st];
            r2[t] += r2[t + st]; r3[t] += r3[t + st];
        }
        __syncthreads();
    }
    const float inv = 1.0f / (float)N_;
    const float muZ = r0[0] * inv, muR = r2[0] * inv;
    const float isZ = rsqrtf(r1[0] * inv - muZ * muZ + EPSV);
    const float isR = rsqrtf(r3[0] * inv - muR * muR + EPSV);

    for (int c = t; c < N_; c += 256) {
        float zn = (pz[c] - muZ) * isZ * gz[c] + bz[c];
        float zv = 1.0f / (1.0f + __expf(-zn));
        float rn = (pr[c] - muR) * isR * gr[c] + br[c];
        float rv = 1.0f / (1.0f + __expf(-rn));
        pz[c]  = zv;                        // z overwrites pre_z (own elements only)
        prh[c] = (bf16)(rv * ph[c]);
    }
}

// ---------------- host side ----------------
extern "C" void kernel_launch(void* const* d_in, const int* in_sizes, int n_in,
                              void* d_out, int out_size, void* d_ws, size_t ws_size,
                              hipStream_t stream)
{
    const float* x   = (const float*)d_in[0];
    const float* hp  = (const float*)d_in[1];
    const float* Wz  = (const float*)d_in[2];
    const float* bz  = (const float*)d_in[3];
    const float* Uz  = (const float*)d_in[4];
    const float* Wr  = (const float*)d_in[5];
    const float* br  = (const float*)d_in[6];
    const float* Ur  = (const float*)d_in[7];
    const float* Wh  = (const float*)d_in[8];
    const float* bh  = (const float*)d_in[9];
    const float* Uh  = (const float*)d_in[10];
    const float* gz  = (const float*)d_in[11];
    const float* bez = (const float*)d_in[12];
    const float* gr  = (const float*)d_in[13];
    const float* ber = (const float*)d_in[14];
    float* out = (float*)d_out;

    char* ws = (char*)d_ws;
    const size_t MB = 1ull << 20;
    bf16*  xb   = (bf16*)(ws + 0);          //  32 MB  bf16 x        (8192x2048)
    bf16*  hb   = (bf16*)(ws + 32 * MB);    //  32 MB  bf16 h_prev
    bf16*  Wzb  = (bf16*)(ws + 64 * MB);    //   8 MB each weight
    bf16*  Uzb  = (bf16*)(ws + 72 * MB);
    bf16*  Wrb  = (bf16*)(ws + 80 * MB);
    bf16*  Urb  = (bf16*)(ws + 88 * MB);
    bf16*  Whb  = (bf16*)(ws + 96 * MB);
    bf16*  Uhb  = (bf16*)(ws + 104 * MB);
    float* preZ = (float*)(ws + 112 * MB);  //  64 MB  pre_z -> z
    float* preR = (float*)(ws + 176 * MB);  //  64 MB  pre_r
    bf16*  rhb  = (bf16*)(ws + 240 * MB);   //  32 MB  bf16 (r*h)

    const int nXH = M_ * K_;                // 16,777,216
    const int nW  = N_ * K_;                //  4,194,304

    cast_f32_bf16<<<nXH / 1024, 256, 0, stream>>>(x,  xb,  nXH);
    cast_f32_bf16<<<nXH / 1024, 256, 0, stream>>>(hp, hb,  nXH);
    cast_f32_bf16<<<nW  / 1024, 256, 0, stream>>>(Wz, Wzb, nW);
    cast_f32_bf16<<<nW  / 1024, 256, 0, stream>>>(Uz, Uzb, nW);
    cast_f32_bf16<<<nW  / 1024, 256, 0, stream>>>(Wr, Wrb, nW);
    cast_f32_bf16<<<nW  / 1024, 256, 0, stream>>>(Ur, Urb, nW);
    cast_f32_bf16<<<nW  / 1024, 256, 0, stream>>>(Wh, Whb, nW);
    cast_f32_bf16<<<nW  / 1024, 256, 0, stream>>>(Uh, Uhb, nW);

    dim3 grid(N_ / 128, M_ / 256);          // 16 x 32 workgroups
    gru_gemm<0><<<grid, 256, 0, stream>>>(xb, Wzb, hb, Uzb, bz, nullptr, nullptr, preZ);
    gru_gemm<0><<<grid, 256, 0, stream>>>(xb, Wrb, hb, Urb, br, nullptr, nullptr, preR);
    ln_sig_kernel<<<M_, 256, 0, stream>>>(preZ, preR, hp, gz, bez, gr, ber, rhb);
    gru_gemm<1><<<grid, 256, 0, stream>>>(xb, Whb, rhb, Uhb, bh, preZ, hp, out);

    (void)in_sizes; (void)n_in; (void)out_size; (void)ws_size;
}
